// OnlineTripletLoss_61907658605016
// MI455X (gfx1250) — matrix operand order repaired and checked by reading, not compiled, not run on hardware
//
#include <hip/hip_runtime.h>

// ---------------------------------------------------------------------------
// OnlineTripletLoss for MI455X (gfx1250, wave32, WMMA + TDM)
//
//   prep:          sq[i] = ||E_i||^2 (f32), Eh = f16(E), init reduction bufs
//   dist_pass<1>:  fused GEMM(Eh Eh^T) + dist epilogue -> hp, hn_fallback
//   dist_pass<2>:  recompute dist, using hp -> hn_semi + global semi-hard count
//   finalize:      resolve global branch, loss = sum(relu)/max(nvalid,1)
//
// dist_pass tiling: grid (32 row-blocks x 8 col-chunks), 256 threads = 8 waves.
// Each wave owns 16 anchor rows with full-K A fragments resident in 128 VGPRs.
// B tiles (16 cols x 512 K = 16 KB) are staged in LDS by the Tensor Data Mover
// (double-buffered, s_wait_tensorcnt) and shared by all 8 waves, with TDM LDS
// padding (16 B per 1 KB row) to kill ds_load_b128 bank conflicts.
// ---------------------------------------------------------------------------

typedef __attribute__((ext_vector_type(16))) _Float16 v16h;
typedef __attribute__((ext_vector_type(8)))  _Float16 v8h;
typedef __attribute__((ext_vector_type(8)))  float    v8f;
typedef __attribute__((ext_vector_type(4)))  unsigned int v4u;
typedef __attribute__((ext_vector_type(8)))  int      v8i;
typedef __attribute__((ext_vector_type(4)))  int      v4i;

#define N_EMB   4096
#define DIMK    512
#define MARGINF 0.3f
#define BIGF    1.0e6f
#define LDS_ROW 520              // 512 halfs + 8 halfs (16 B) pad -> 1040 B stride

#if __has_builtin(__builtin_amdgcn_tensor_load_to_lds) && \
    __has_builtin(__builtin_amdgcn_s_wait_tensorcnt)
#define USE_TDM 1
#else
#define USE_TDM 0
#endif

// ---------------------------------------------------------------------------
// prep: one 32-lane group per row. f32 -> f16 copy, squared norms, buffer init.
// ---------------------------------------------------------------------------
__global__ void triplet_prep_kernel(const float* __restrict__ E,
                                    _Float16* __restrict__ Eh,
                                    float* __restrict__ sq,
                                    float* __restrict__ hp,
                                    float* __restrict__ hnf,
                                    float* __restrict__ hs,
                                    unsigned* __restrict__ scnt)
{
    const int gt   = blockIdx.x * blockDim.x + threadIdx.x;
    const int row  = gt >> 5;
    const int lane = gt & 31;
    if (row >= N_EMB) return;

    const float* __restrict__ r = E + (size_t)row * DIMK;
    _Float16* __restrict__ w    = Eh + (size_t)row * DIMK;

    float acc = 0.0f;
    #pragma unroll
    for (int i = 0; i < DIMK / 32; ++i) {
        const int c = lane + i * 32;
        const float v = r[c];
        acc += v * v;
        w[c] = (_Float16)v;
    }
    for (int m = 16; m >= 1; m >>= 1) acc += __shfl_xor(acc, m);

    if (lane == 0) {
        sq[row]  = acc;
        hp[row]  = 0.0f;       // max identity (dists >= 0)
        hnf[row] = 0.0f;
        hs[row]  = 3.0e38f;    // min identity
        if (row == 0) *scnt = 0u;
    }
}

#if USE_TDM
// Issue a TDM load of one B tile: 16 contiguous rows x 512 halfs from Eh into
// LDS at ldsByteOff, with a 16 B pad after every 1024 B row (LDS_ROW stride).
// 6-arg builtin form (clang-23 / therock-10.0 lane):
//   (uint32x4 g0, int32x8 g1, int32x4 g2, int32x4 g3, int32x8, i32 cpol)
__device__ __forceinline__ void tdm_load_tile(const _Float16* gsrc,
                                              unsigned ldsByteOff)
{
    const unsigned long long ga = (unsigned long long)(uintptr_t)gsrc;
    v4u g0;
    g0[0] = 1u;                                              // count=1 (valid), user
    g0[1] = ldsByteOff;                                      // lds_addr (bytes)
    g0[2] = (unsigned)(ga & 0xFFFFFFFFull);                  // global_addr[31:0]
    g0[3] = (unsigned)((ga >> 32) & 0x01FFFFFFull)           // global_addr[56:32]
          | (2u << 30);                                      // type = 2 ("image")
    v8i g1;
    g1[0] = (int)((1u << 16)                                 // data_size = 2 B
                | (1u << 20)                                 // pad_enable
                | (7u << 22)                                 // pad_interval = 256 dw
                | (3u << 25));                               // pad_amount  = 4 dw
    g1[1] = (int)((512u & 0xFFFFu) << 16);                   // tensor_dim0 lo16
    g1[2] = (int)(((512u >> 16) & 0xFFFFu)                   // tensor_dim0 hi16
                | ((4096u & 0xFFFFu) << 16));                // tensor_dim1 lo16
    g1[3] = (int)(((4096u >> 16) & 0xFFFFu)                  // tensor_dim1 hi16
                | (512u << 16));                             // tile_dim0 = 512
    g1[4] = 16;                                              // tile_dim1=16, tile_dim2=0
    g1[5] = 512;                                             // tensor_dim0_stride lo32
    g1[6] = 0;                                               // stride0 hi | stride1 lo
    g1[7] = 0;                                               // stride1 hi
    const v4i z4 = {0, 0, 0, 0};                             // groups 2/3: 2D tensor
    const v8i z8 = {0, 0, 0, 0, 0, 0, 0, 0};
    __builtin_amdgcn_tensor_load_to_lds(g0, g1, z4, z4, z8, 0);
}
#endif

// ---------------------------------------------------------------------------
// fused GEMM + distance epilogue.
// ---------------------------------------------------------------------------
template <int PASS>
__global__ __launch_bounds__(256)
void triplet_dist_pass_kernel(const _Float16* __restrict__ Eh,
                              const float* __restrict__ sq,
                              const int* __restrict__ labels,
                              float* __restrict__ hp,     // PASS1: out, PASS2: in
                              float* __restrict__ hnf,    // PASS1: out
                              float* __restrict__ hs,     // PASS2: out
                              unsigned* __restrict__ scnt)// PASS2: out
{
    __shared__ _Float16 smem[2][16 * LDS_ROW];   // 2 x 16.25 KB B-tile buffers

    const int lane = threadIdx.x & 31;
    const int wave = threadIdx.x >> 5;
    const int hi   = lane >> 4;      // lane half: 0 or 1
    const int mlo  = lane & 15;
    const int rowBase = blockIdx.x * 128 + wave * 16;   // 32 row-blocks
    const int ct0     = blockIdx.y * 32;                // 8 col-chunks x 32 tiles

    // ---- A fragments (ISA 16-bit A 16x32 layout): M = lane&15,
    //      v0..3 hold K = kk + hi*8 + 0..7, v4..7 hold K = kk + 16 + hi*8 + 0..7
    const _Float16* __restrict__ Arow = Eh + (size_t)(rowBase + mlo) * DIMK;
    v16h afrag[DIMK / 32];
    #pragma unroll
    for (int s = 0; s < DIMK / 32; ++s) {
        const int kk = s * 32;
        const v8h a0 = *(const v8h*)(Arow + kk + hi * 8);
        const v8h a1 = *(const v8h*)(Arow + kk + 16 + hi * 8);
        v16h a;
        #pragma unroll
        for (int i = 0; i < 8; ++i) { a[i] = a0[i]; a[i + 8] = a1[i]; }
        afrag[s] = a;
    }

    // ---- per-lane anchor-row metadata in C-layout order: gRow(r) = rowBase + r + hi*8
    float sqA[8]; int labA[8]; float hpA[8];
    #pragma unroll
    for (int r = 0; r < 8; ++r) {
        const int gr = rowBase + r + hi * 8;
        sqA[r]  = sq[gr];
        labA[r] = labels[gr];
        hpA[r]  = (PASS == 2) ? hp[gr] : 0.0f;
    }

    float red0[8], red1[8];  // PASS1: hp / hnf running max. PASS2: hn_semi running min.
    #pragma unroll
    for (int r = 0; r < 8; ++r) {
        red0[r] = (PASS == 1) ? 0.0f : 3.0e38f;
        red1[r] = 0.0f;
    }
    unsigned semiCount = 0u;

#if USE_TDM
    if (wave == 0) {   // wave-uniform: one TDM issue per call, EXEC ignored
        tdm_load_tile(Eh + (size_t)(ct0 + 0) * 16 * DIMK,
                      (unsigned)(uintptr_t)(void*)&smem[0][0]);
        tdm_load_tile(Eh + (size_t)(ct0 + 1) * 16 * DIMK,
                      (unsigned)(uintptr_t)(void*)&smem[1][0]);
    }
#endif

    for (int i = 0; i < 32; ++i) {
        const int ct   = ct0 + i;
        const int gcol = ct * 16 + mlo;

#if USE_TDM
        if (wave == 0) {
            if (i < 31) __builtin_amdgcn_s_wait_tensorcnt(1); // tile i landed
            else        __builtin_amdgcn_s_wait_tensorcnt(0); // last tile
        }
        __syncthreads();
#else
        __syncthreads();   // previous readers of smem[i&1] are done
        {   // cooperative copy: 256 threads x 64 B = 16 KB tile (padded rows)
            const int t = threadIdx.x, rr = t >> 4, part = t & 15;
            const _Float16* src = Eh + (size_t)(ct * 16 + rr) * DIMK + part * 32;
            _Float16* dst = &smem[i & 1][rr * LDS_ROW + part * 32];
            #pragma unroll
            for (int q = 0; q < 4; ++q)
                *(v8h*)(dst + q * 8) = *(const v8h*)(src + q * 8);
        }
        __syncthreads();
#endif

        const float sqB = sq[gcol];
        const int  labB = labels[gcol];
        const _Float16* bb = &smem[i & 1][mlo * LDS_ROW];

        v8f acc = {};
        #pragma unroll
        for (int s = 0; s < DIMK / 32; ++s) {
            const int kk = s * 32;
            // B (32x16): N = lane&15, elements = K kk + hi*8 + 0..7, kk + 16 + hi*8 + 0..7
            const v8h b0 = *(const v8h*)(bb + kk + hi * 8);
            const v8h b1 = *(const v8h*)(bb + kk + 16 + hi * 8);
            v16h b;
            #pragma unroll
            for (int q = 0; q < 8; ++q) { b[q] = b0[q]; b[q + 8] = b1[q]; }
            acc = __builtin_amdgcn_wmma_f32_16x16x32_f16(
                      false, afrag[s], false, b, (short)0, acc, false, false);
        }

        // epilogue: C layout -> dist -> masked reductions
        #pragma unroll
        for (int r = 0; r < 8; ++r) {
            const int gr = rowBase + r + hi * 8;
            const float d2 = sqA[r] + sqB - 2.0f * acc[r];
            float dist = sqrtf(fmaxf(d2, 1e-12f));
            if (gcol == gr) dist = 1e-6f;   // exact zero diagonal (ref: sqrt(1e-12))
            const bool pos = (labB == labA[r]);
            if (PASS == 1) {
                if (pos) red0[r] = fmaxf(red0[r], dist);   // hardest positive
                else     red1[r] = fmaxf(red1[r], dist);   // farthest negative
            } else {
                const bool semi = (!pos) && (dist > hpA[r]) && (dist < hpA[r] + MARGINF);
                red0[r] = fminf(red0[r], semi ? dist : dist + BIGF);
                semiCount += semi ? 1u : 0u;
            }
        }

#if USE_TDM
        __syncthreads();   // all waves done reading smem[i&1] before refill
        if (wave == 0 && i + 2 < 32)
            tdm_load_tile(Eh + (size_t)(ct0 + i + 2) * 16 * DIMK,
                          (unsigned)(uintptr_t)(void*)&smem[i & 1][0]);
#endif
    }

    // reduce across the 16 lanes of each half-wave (masks 1,2,4,8 stay in-half)
    #pragma unroll
    for (int r = 0; r < 8; ++r) {
        const int gr = rowBase + r + hi * 8;
        if (PASS == 1) {
            float v0 = red0[r], v1 = red1[r];
            for (int m = 8; m >= 1; m >>= 1) {
                v0 = fmaxf(v0, __shfl_xor(v0, m));
                v1 = fmaxf(v1, __shfl_xor(v1, m));
            }
            if (mlo == 0) {   // non-negative floats: uint order == float order
                atomicMax((unsigned*)&hp[gr],  __float_as_uint(v0));
                atomicMax((unsigned*)&hnf[gr], __float_as_uint(v1));
            }
        } else {
            float v0 = red0[r];
            for (int m = 8; m >= 1; m >>= 1)
                v0 = fminf(v0, __shfl_xor(v0, m));
            if (mlo == 0)
                atomicMin((unsigned*)&hs[gr], __float_as_uint(v0));
        }
    }

    if (PASS == 2) {
        int c = (int)semiCount;
        for (int m = 16; m >= 1; m >>= 1) c += __shfl_xor(c, m);
        if (lane == 0 && c > 0) atomicAdd(scnt, (unsigned)c);
    }
}

// ---------------------------------------------------------------------------
// finalize: resolve global semi-hard branch, produce scalar loss.
// ---------------------------------------------------------------------------
__global__ void triplet_finalize_kernel(const float* __restrict__ hp,
                                        const float* __restrict__ hnf,
                                        const float* __restrict__ hs,
                                        const unsigned* __restrict__ scnt,
                                        float* __restrict__ out)
{
    __shared__ float s_sum[256];
    __shared__ float s_cnt[256];
    const int t = threadIdx.x;
    const bool hasSemi = (*scnt) > 0u;

    float sum = 0.0f, cnt = 0.0f;
    for (int i = t; i < N_EMB; i += 256) {
        const float hn = hasSemi ? hs[i] : hnf[i];
        const float tr = fmaxf(hp[i] - hn + MARGINF, 0.0f);
        sum += tr;
        cnt += (tr > 0.0f) ? 1.0f : 0.0f;
    }
    s_sum[t] = sum; s_cnt[t] = cnt;
    __syncthreads();
    for (int o = 128; o > 0; o >>= 1) {
        if (t < o) { s_sum[t] += s_sum[t + o]; s_cnt[t] += s_cnt[t + o]; }
        __syncthreads();
    }
    if (t == 0) out[0] = s_sum[0] / fmaxf(s_cnt[0], 1.0f);
}

// ---------------------------------------------------------------------------
extern "C" void kernel_launch(void* const* d_in, const int* in_sizes, int n_in,
                              void* d_out, int out_size, void* d_ws, size_t ws_size,
                              hipStream_t stream)
{
    const float* E      = (const float*)d_in[0];   // [4096, 512] f32
    const int*   labels = (const int*)d_in[1];     // [4096] int
    float* out = (float*)d_out;

    char* ws = (char*)d_ws;
    _Float16* Eh = (_Float16*)ws;
    size_t off = (size_t)N_EMB * DIMK * sizeof(_Float16);      // 4 MB
    float* sq  = (float*)(ws + off); off += N_EMB * sizeof(float);
    float* hp  = (float*)(ws + off); off += N_EMB * sizeof(float);
    float* hnf = (float*)(ws + off); off += N_EMB * sizeof(float);
    float* hs  = (float*)(ws + off); off += N_EMB * sizeof(float);
    unsigned* scnt = (unsigned*)(ws + off);

    triplet_prep_kernel<<<(N_EMB * 32) / 256, 256, 0, stream>>>(
        E, Eh, sq, hp, hnf, hs, scnt);

    dim3 grid(N_EMB / 128, 8);   // 32 row-blocks x 8 column-chunks = 256 WGs
    triplet_dist_pass_kernel<1><<<grid, 256, 0, stream>>>(
        Eh, sq, labels, hp, hnf, hs, scnt);
    triplet_dist_pass_kernel<2><<<grid, 256, 0, stream>>>(
        Eh, sq, labels, hp, hnf, hs, scnt);

    triplet_finalize_kernel<<<1, 256, 0, stream>>>(hp, hnf, hs, scnt, out);
}
